// BaseAttention_58411555226335
// MI455X (gfx1250) — matrix operand order repaired and checked by reading, not compiled, not run on hardware
//
#include <hip/hip_runtime.h>
#include <math.h>

typedef _Float16 half_t;
typedef __attribute__((ext_vector_type(16))) _Float16 v16h;
typedef __attribute__((ext_vector_type(8)))  _Float16 v8h;
typedef __attribute__((ext_vector_type(8)))  float    v8f;

#define BATCH     32768
#define OBS_DIM   576
#define BEGIN_C   32
#define NUM_OBJS  32
#define D_MODEL   128
#define OUT_DIM   192          // 64 aux + 128 attention output
#define ROWS_PER_BLOCK 4       // 4 batch rows -> 128 tokens per workgroup
#define TOKENS    (ROWS_PER_BLOCK * NUM_OBJS)   // 128

// ---- LDS layout ----
// halves: sA1 128*32, sW1 128*32, sW2 128*128, sW3 128*128, sHa 128*128, sHb 128*128
// floats: mask 128, msum 4, query 512, p 512, logits 128, wts 128
#define SMEM_HALFS  (128*32*2 + 128*128*4)
#define SMEM_FLOATS (128 + 4 + 512 + 512 + 128 + 128)
#define SMEM_BYTES  (SMEM_HALFS*2 + SMEM_FLOATS*4)

// gfx1250 async global->LDS path (ASYNCcnt), guarded for toolchain portability.
#if __has_builtin(__builtin_amdgcn_global_load_async_to_lds_b128) && \
    __has_builtin(__builtin_amdgcn_s_wait_asynccnt)
#define USE_ASYNC_LDS 1
#else
#define USE_ASYNC_LDS 0
#endif

typedef int int4v __attribute__((vector_size(16)));
typedef int4v __attribute__((address_space(1))) * gbl_i4p;   // non-const per builtin proto
typedef int4v __attribute__((address_space(3))) * lds_i4p;

__device__ __forceinline__ void copy16_to_lds(const void* g, void* l) {
#if USE_ASYNC_LDS
    void* gnc = const_cast<void*>(g);
    __builtin_amdgcn_global_load_async_to_lds_b128((gbl_i4p)gnc, (lds_i4p)l, 0, 0);
#else
    *(uint4*)l = *(const uint4*)g;
#endif
}

__device__ __forceinline__ void async_stage_fence() {
#if USE_ASYNC_LDS
    __builtin_amdgcn_s_wait_asynccnt(0);
#endif
}

// Prep (one tiny launch, deterministic each call):
//  - fp16 row-major weights in workspace, W1 zero-padded 15 -> 32 on K
//  - Mt[k][e] = sum_d Uq[d][e] * Ur[d][k]   (fuses q = query Uq^T and p = Ur^T q,
//    since logits = (Ur^T Uq query) . x_real ; Mt is batch-independent)
__global__ void prep_weights(const float* __restrict__ W1,
                             const float* __restrict__ W2,
                             const float* __restrict__ W3,
                             const float* __restrict__ Uq,
                             const float* __restrict__ Ur,
                             half_t* __restrict__ w1h,
                             half_t* __restrict__ w2h,
                             half_t* __restrict__ w3h,
                             float*  __restrict__ Mt) {
    int i = blockIdx.x * blockDim.x + threadIdx.x;   // 16384 threads
    if (i < 128 * 32) {
        int r = i >> 5, c = i & 31;
        w1h[i] = (half_t)(c < 15 ? W1[r * 15 + c] : 0.0f);
    }
    if (i < 128 * 128) {
        w2h[i] = (half_t)W2[i];
        w3h[i] = (half_t)W3[i];
        const int k = i >> 7, e = i & 127;
        float s = 0.0f;
        for (int d = 0; d < 128; ++d) s += Uq[d * 128 + e] * Ur[d * 128 + k];
        Mt[i] = s;
    }
}

// One MLP layer on a 128-token tile: D = act(A[128xK] * W^T[Kx128] + b), optionally * mask.
// Wave `wave` owns M-tile rows [wave*16, wave*16+16). Emits 8*(K/32) v_wmma per wave.
__device__ __forceinline__ void mlp_layer(const half_t* __restrict__ sIn, int Kdim,
                                          const half_t* __restrict__ sW,
                                          const float* __restrict__ bias,
                                          half_t* __restrict__ sOut,
                                          const float* __restrict__ sMask,
                                          bool relu, bool domask,
                                          int wave, int lane) {
    v8f acc[8] = {};
    const int mlo = lane & 15;          // A row / B column / D column within tile
    const int hsel = lane >> 4;         // lane-half select per ISA fragment layout
    const half_t* aRow = sIn + (wave * 16 + mlo) * Kdim;

    for (int k0 = 0; k0 < Kdim; k0 += 32) {
        // A fragment (16x32 f16): elems 0-15 = K[k0+16h .. +15]
        union { v16h v; v8h h[2]; } a;
        const half_t* ap = aRow + k0 + hsel * 16;
        a.h[0] = *(const v8h*)ap;
        a.h[1] = *(const v8h*)(ap + 8);
#pragma unroll
        for (int j = 0; j < 8; ++j) {
            // B fragment (32x16): lane holds col n=mlo, K range k0+16*hsel .. +15.
            // B^T[n][k] == W[n][k] row-major -> one contiguous 32B LDS read.
            const v16h b = *(const v16h*)(sW + (j * 16 + mlo) * Kdim + k0 + hsel * 16);
            acc[j] = __builtin_amdgcn_wmma_f32_16x16x32_f16(
                false, a.v, false, b, (short)0, acc[j], false, false);
        }
    }
#pragma unroll
    for (int j = 0; j < 8; ++j) {
        const int col = j * 16 + mlo;
        const float bv = bias[col];
#pragma unroll
        for (int r = 0; r < 8; ++r) {
            const int row = wave * 16 + r + 8 * hsel;   // D layout: VGPR r -> M = r + 8*hsel
            float v = acc[j][r] + bv;
            if (relu) v = fmaxf(v, 0.0f);
            if (domask) v *= sMask[row];
            sOut[row * D_MODEL + col] = (half_t)v;
        }
    }
}

__global__ void __launch_bounds__(256)
attn_kernel(const float* __restrict__ obs,
            const half_t* __restrict__ w1h,   // contiguous with w2h, w3h in workspace
            const float* __restrict__ b1,
            const float* __restrict__ b2,
            const float* __restrict__ b3,
            const float* __restrict__ Mt,
            float* __restrict__ out) {
    extern __shared__ char smem_raw[];
    half_t* sA1 = (half_t*)smem_raw;        // [128][32]  layer-1 input (padded)
    half_t* sW1 = sA1 + 128 * 32;           // [128][32]  weight image part 1
    half_t* sW2 = sW1 + 128 * 32;           // [128][128] weight image part 2
    half_t* sW3 = sW2 + 128 * 128;          // [128][128] weight image part 3
    half_t* sHa = sW3 + 128 * 128;          // [128][128] activations ping
    half_t* sHb = sHa + 128 * 128;          // [128][128] activations pong (also obs scratch)
    float* fb     = (float*)(sHb + 128 * 128);
    float* sMask  = fb;                     // [128]
    float* sMsum  = sMask + 128;            // [4]
    float* sQuery = sMsum + 4;              // [4][128]
    float* sP     = sQuery + 512;           // [4][128]
    float* sLogit = sP + 512;               // [128]
    float* sWt    = sLogit + 128;           // [128]

    const int tid  = threadIdx.x;
    const int wave = tid >> 5;
    const int lane = tid & 31;
    const int b0   = blockIdx.x * ROWS_PER_BLOCK;

    __builtin_prefetch(Mt + tid * 16, 0, 1);   // global_prefetch_b8: keep Mt hot in GL2

    // ---- Stage weight image (72KB, contiguous) global -> LDS via async-to-LDS ----
    {
        const char* wsrc = (const char*)w1h;   // w1h|w2h|w3h contiguous in workspace
        char*       wdst = (char*)sW1;         // sW1|sW2|sW3 contiguous in LDS
        for (int i = tid; i < 4608; i += 256)
            copy16_to_lds(wsrc + (size_t)i * 16, wdst + i * 16);
    }
    // ---- Stage obs attention slab (4 x 512 f32) into sHb scratch (dead until layer 2) ----
    {
        float* scr = (float*)sHb;
        for (int i = tid; i < 512; i += 256) {
            const int bl = i >> 7, ii = i & 127;
            copy16_to_lds(obs + (size_t)(b0 + bl) * OBS_DIM + BEGIN_C + ii * 4,
                          scr + i * 4);
        }
    }
    // ---- Zero layer-1 input tile (K padding 15..31 must be 0) ----
    for (int i = tid; i < 1024; i += 256) ((uint2*)sA1)[i] = make_uint2(0u, 0u);
    {   // aux passthrough while async copies fly: out[:,0:32]=obs[:,0:32]; out[:,32:64]=obs[:,544:576]
        const int bl = tid >> 6, jj = tid & 63;   // 4*64 == 256 == blockDim
        const size_t src = (size_t)(b0 + bl) * OBS_DIM + (jj < 32 ? jj : 512 + jj);
        out[(size_t)(b0 + bl) * OUT_DIM + jj] = obs[src];
    }
    async_stage_fence();     // s_wait_asynccnt 0
    __syncthreads();

    // ---- Convert obs slab: feats -> sA1 fp16, mask channel -> sMask ----
    {
        const float* scr = (const float*)sHb;
        for (int i = tid; i < ROWS_PER_BLOCK * 512; i += 256) {
            const int bl = i >> 9, ii = i & 511;
            const float v = scr[i];
            const int n = ii >> 4, f = ii & 15, t = bl * NUM_OBJS + n;
            if (f == 15) sMask[t] = v;
            else         sA1[t * 32 + f] = (half_t)v;
        }
    }
    __syncthreads();

    if (tid < ROWS_PER_BLOCK) {
        float s = 0.0f;
        for (int n = 0; n < NUM_OBJS; ++n) s += sMask[tid * NUM_OBJS + n];
        sMsum[tid] = s;
    }

    // ---- 3-layer MLP over 128 tokens via WMMA ----
    mlp_layer(sA1, 32,  sW1, b1, sHa, sMask, true,  false, wave, lane);
    __syncthreads();
    mlp_layer(sHa, 128, sW2, b2, sHb, sMask, true,  false, wave, lane);
    __syncthreads();
    mlp_layer(sHb, 128, sW3, b3, sHa, sMask, false, true,  wave, lane);  // sHa = x_real (masked)
    __syncthreads();

    // ---- query = masked mean of x_real over objects ----
    for (int i = tid; i < ROWS_PER_BLOCK * D_MODEL; i += 256) {
        const int bl = i >> 7, d = i & 127;
        const half_t* xr = sHa + (bl * NUM_OBJS) * D_MODEL + d;
        float s = 0.0f;
#pragma unroll 8
        for (int n = 0; n < NUM_OBJS; ++n) s += (float)xr[n * D_MODEL];
        sQuery[i] = s / (sMsum[bl] + 1e-5f);
    }
    __syncthreads();
    // ---- p[bl] = query[bl] @ Mt^T  (Mt = (Uq^T Ur)^T, precomputed; logits = p . x_real) ----
    for (int i = tid; i < ROWS_PER_BLOCK * D_MODEL; i += 256) {
        const int bl = i >> 7, k = i & 127;
        const float* qr = sQuery + bl * D_MODEL;
        const float* mr = Mt + k * D_MODEL;      // contiguous 512B row, GL2-resident
        float s = 0.0f;
#pragma unroll 4
        for (int e = 0; e < D_MODEL; ++e) s += qr[e] * mr[e];
        sP[i] = s;
    }
    __syncthreads();
    // ---- logits ----
    if (tid < TOKENS) {
        const int bl = tid >> 5;
        const half_t* xr = sHa + tid * D_MODEL;
        const float* p = sP + bl * D_MODEL;
        float s = 0.0f;
        for (int k = 0; k < D_MODEL; ++k) s += (float)xr[k] * p[k];
        sLogit[tid] = s + (1.0f - sMask[tid]) * -1.0e9f;
    }
    __syncthreads();
    // ---- softmax over 32 objects per row ----
    if (tid < ROWS_PER_BLOCK) {
        float mx = -INFINITY;
        for (int n = 0; n < NUM_OBJS; ++n) mx = fmaxf(mx, sLogit[tid * NUM_OBJS + n]);
        float sum = 0.0f;
        for (int n = 0; n < NUM_OBJS; ++n) {
            const float e = __expf(sLogit[tid * NUM_OBJS + n] - mx);
            sWt[tid * NUM_OBJS + n] = e;
            sum += e;
        }
        const float inv = 1.0f / sum;
        for (int n = 0; n < NUM_OBJS; ++n) sWt[tid * NUM_OBJS + n] *= inv;
    }
    __syncthreads();
    // ---- out_att = w . x_real ----
    for (int i = tid; i < ROWS_PER_BLOCK * D_MODEL; i += 256) {
        const int bl = i >> 7, d = i & 127;
        const half_t* xr = sHa + (bl * NUM_OBJS) * D_MODEL + d;
        const float* wv = sWt + bl * NUM_OBJS;
        float s = 0.0f;
#pragma unroll 8
        for (int n = 0; n < NUM_OBJS; ++n) s += wv[n] * (float)xr[n * D_MODEL];
        out[(size_t)(b0 + bl) * OUT_DIM + 64 + d] = s;
    }
}

extern "C" void kernel_launch(void* const* d_in, const int* in_sizes, int n_in,
                              void* d_out, int out_size, void* d_ws, size_t ws_size,
                              hipStream_t stream) {
    const float* obs = (const float*)d_in[0];
    const float* W1  = (const float*)d_in[1];
    const float* b1  = (const float*)d_in[2];
    const float* W2  = (const float*)d_in[3];
    const float* b2  = (const float*)d_in[4];
    const float* W3  = (const float*)d_in[5];
    const float* b3  = (const float*)d_in[6];
    const float* Uq  = (const float*)d_in[7];
    const float* Ur  = (const float*)d_in[8];
    float* out = (float*)d_out;

    half_t* w1h = (half_t*)d_ws;            // 128*32  (contiguous weight image begins)
    half_t* w2h = w1h + 128 * 32;           // 128*128
    half_t* w3h = w2h + 128 * 128;          // 128*128 (weight image = 73,728 B total)
    float*  Mt  = (float*)(w3h + 128 * 128);// 128*128 f32 (Uq^T Ur)^T

    prep_weights<<<64, 256, 0, stream>>>(W1, W2, W3, Uq, Ur, w1h, w2h, w3h, Mt);

    (void)hipFuncSetAttribute((const void*)attn_kernel,
                              hipFuncAttributeMaxDynamicSharedMemorySize,
                              (int)SMEM_BYTES);
    attn_kernel<<<BATCH / ROWS_PER_BLOCK, 256, SMEM_BYTES, stream>>>(
        obs, w1h, b1, b2, b3, Mt, out);
    (void)in_sizes; (void)n_in; (void)out_size; (void)ws_size;
}
